// AttDGCNNEncoder_18227841204610
// MI455X (gfx1250) — matrix operand order repaired and checked by reading, not compiled, not run on hardware
//
#include <hip/hip_runtime.h>
#include <hip/hip_bf16.h>

typedef __attribute__((ext_vector_type(16))) _Float16 v16h;
typedef __attribute__((ext_vector_type(2)))  _Float16 v2h;
typedef __attribute__((ext_vector_type(8)))  float    v8f;

#define BG 32
#define NN 2048
#define KNB 8
#define BN_EPS 1e-5f

// ---------------------------------------------------------------------------
// WMMA fragment-order LDS staging helpers.
// A 16x32 f16 fragment: element (m, klocal) -> lane = m + 16*(klocal in
// [8,16)u[24,32)), half = (klocal&7) + 8*(klocal>=16).
// B 32x16 f16 fragment: element (klocal, n) -> lane = n + 16*(klocal in
// [8,16)u[24,32)), half = (klocal&7) + 8*(klocal>=16).  Pairs (even klocal, +1) stay within one lane => b32 store.
// Each lane then reads its 16 contiguous halves (32B) => ds_load_b128 x2.
// ---------------------------------------------------------------------------
__device__ __forceinline__ void put_a_pair(_Float16* frag, int m, int klocal,
                                           _Float16 v0, _Float16 v1)
{
    int lane = m + ((klocal & 8) ? 16 : 0);
    int h = (klocal & 7) + ((klocal & 16) ? 8 : 0);
    *(v2h*)(frag + lane * 16 + h) = (v2h){v0, v1};
}
__device__ __forceinline__ void put_b_pair(_Float16* frag, int n, int klocal,
                                           _Float16 v0, _Float16 v1)
{
    int lane = n + ((klocal & 8) ? 16 : 0);
    int h = (klocal & 7) + ((klocal & 16) ? 8 : 0);
    *(v2h*)(frag + lane * 16 + h) = (v2h){v0, v1};
}

// ---------------------------------------------------------------------------
// row squared-norms
// ---------------------------------------------------------------------------
__global__ void rowsq_kernel(const float* __restrict__ x, int ldx, int cin,
                             float* __restrict__ x2, int nrows)
{
    int i = blockIdx.x * blockDim.x + threadIdx.x;
    if (i >= nrows) return;
    float s = 0.f;
    for (int c = 0; c < cin; ++c) { float v = x[(size_t)i * ldx + c]; s += v * v; }
    x2[i] = s;
}

// ---------------------------------------------------------------------------
// kNN: 16-query strip per block (4 waves). Gram via WMMA f16, distance
// d = |xi|^2 + |xj|^2 - 2 xi.xj, register top-8 per row, 8-way merge.
// ---------------------------------------------------------------------------
template<int KC, int CIN>
__global__ __launch_bounds__(128) void knn_kernel(
    const float* __restrict__ x, int ldx,
    const float* __restrict__ x2, int* __restrict__ idx)
{
    constexpr int KP = KC * 32;
    __shared__ __align__(32) _Float16 Af[KC * 512];        // fragment order
    __shared__ __align__(32) _Float16 Bf[4][KC * 512];     // per-wave fragments
    __shared__ float    Ds[4][16 * 16];
    __shared__ float    x2r[16];
    __shared__ float    Md[128][8];
    __shared__ int      Mi[128][8];

    const int b   = blockIdx.y;
    const int m0  = blockIdx.x * 16;
    const int tid = threadIdx.x;
    const int wv = tid >> 5, lane = tid & 31;
    const size_t rowbase = (size_t)b * NN;

    // stage A tile directly in fragment order (pairs -> 32-bit LDS stores)
    for (int t = tid; t < 16 * KP / 2; t += 128) {
        int m  = t / (KP / 2);
        int kp = (t % (KP / 2)) * 2;
        const float* xr = x + (rowbase + m0 + m) * ldx;
        float v0 = (kp     < CIN) ? xr[kp]     : 0.f;
        float v1 = (kp + 1 < CIN) ? xr[kp + 1] : 0.f;
        put_a_pair(Af + (kp >> 5) * 512, m, kp & 31, (_Float16)v0, (_Float16)v1);
    }
    if (tid < 16) x2r[tid] = x2[rowbase + m0 + tid];
    __syncthreads();

    v16h af[KC];
    #pragma unroll
    for (int c = 0; c < KC; ++c)
        af[c] = *(const v16h*)(Af + c * 512 + lane * 16);

    float td[8]; int ti[8];
    #pragma unroll
    for (int q = 0; q < 8; ++q) { td[q] = 3.4e38f; ti[q] = 0; }

    _Float16* Bw = &Bf[wv][0];
    for (int ct = wv; ct < NN / 16; ct += 4) {
        const int n0 = ct * 16;
        for (int t = lane; t < 16 * KP / 2; t += 32) {
            int n  = t / (KP / 2);
            int kp = (t % (KP / 2)) * 2;
            const float* xr = x + (rowbase + n0 + n) * ldx;
            float v0 = (kp     < CIN) ? xr[kp]     : 0.f;
            float v1 = (kp + 1 < CIN) ? xr[kp + 1] : 0.f;
            put_b_pair(Bw + (kp >> 5) * 512, n, kp & 31, (_Float16)v0, (_Float16)v1);
        }
        __syncthreads();

        v8f acc = {};
        #pragma unroll
        for (int c = 0; c < KC; ++c) {
            v16h bf = *(const v16h*)(Bw + c * 512 + lane * 16);
            acc = __builtin_amdgcn_wmma_f32_16x16x32_f16(
                false, af[c], false, bf, (short)0, acc, false, false);
        }
        {
            const int n = lane & 15, rofs = (lane >> 4) << 3;
            const float x2c = x2[rowbase + n0 + n];
            #pragma unroll
            for (int r = 0; r < 8; ++r) {
                int row = r + rofs;
                Ds[wv][row * 16 + n] = x2r[row] + x2c - 2.f * acc[r];
            }
        }
        __syncthreads();
        {
            const int row = lane >> 1, cb = (lane & 1) * 8;
            #pragma unroll
            for (int j = 0; j < 8; ++j) {
                float d = Ds[wv][row * 16 + cb + j];
                int gi = n0 + cb + j;
                if (d < td[7]) {
                    td[7] = d; ti[7] = gi;
                    #pragma unroll
                    for (int q = 7; q > 0; --q)
                        if (td[q] < td[q - 1]) {
                            float tf = td[q]; td[q] = td[q - 1]; td[q - 1] = tf;
                            int   tt = ti[q]; ti[q] = ti[q - 1]; ti[q - 1] = tt;
                        }
                }
            }
        }
    }

    __syncthreads();
    #pragma unroll
    for (int q = 0; q < 8; ++q) { Md[tid][q] = td[q]; Mi[tid][q] = ti[q]; }
    __syncthreads();

    if (tid < 16) {
        float bd[8]; int bi[8];
        #pragma unroll
        for (int q = 0; q < 8; ++q) { bd[q] = 3.4e38f; bi[q] = 0; }
        for (int w = 0; w < 4; ++w)
            for (int hh = 0; hh < 2; ++hh) {
                int src = w * 32 + tid * 2 + hh;
                #pragma unroll
                for (int j = 0; j < 8; ++j) {
                    float d = Md[src][j]; int gi = Mi[src][j];
                    if (d < bd[7]) {
                        bd[7] = d; bi[7] = gi;
                        #pragma unroll
                        for (int q = 7; q > 0; --q)
                            if (bd[q] < bd[q - 1]) {
                                float tf = bd[q]; bd[q] = bd[q - 1]; bd[q - 1] = tf;
                                int   tt = bi[q]; bi[q] = bi[q - 1]; bi[q - 1] = tt;
                            }
                    }
                }
            }
        #pragma unroll
        for (int q = 0; q < 8; ++q)
            idx[(rowbase + m0 + tid) * KNB + q] = bi[q];
    }
}

// ---------------------------------------------------------------------------
// Edge GEMM: h_pre[e, :] = [xi, xj-xi] @ W + b  via WMMA; gather on the fly.
// ---------------------------------------------------------------------------
template<int CIN, int NW>
__global__ __launch_bounds__(32 * NW) void edge_gemm_kernel(
    const float* __restrict__ x, int ldx,
    const int* __restrict__ idx,
    const float* __restrict__ W, const float* __restrict__ bias,
    int cout, float* __restrict__ out)
{
    constexpr int K2 = 2 * CIN;
    constexpr int KC = (K2 + 31) / 32;
    constexpr int KP = KC * 32;
    __shared__ __align__(32) _Float16 Af[KC * 512];
    __shared__ __align__(32) _Float16 Wf[NW * 512];

    const int tid = threadIdx.x, wv = tid >> 5, lane = tid & 31;
    const int m0 = blockIdx.x * 16;
    const int n0 = blockIdx.y * 16 * NW;

    for (int t = tid; t < 16 * KP / 2; t += 32 * NW) {
        int m  = t / (KP / 2);
        int kp = (t % (KP / 2)) * 2;
        size_t e = (size_t)(m0 + m);
        int bb = (int)(e / (NN * KNB));
        int nn = (int)((e / KNB) % NN);
        int kk = (int)(e % KNB);
        int j  = idx[((size_t)bb * NN + nn) * KNB + kk];
        const float* xi = x + ((size_t)bb * NN + nn) * ldx;
        const float* xj = x + ((size_t)bb * NN + j)  * ldx;
        auto ev = [&](int c) -> float {
            if (c < CIN) return xi[c];
            if (c < K2)  return xj[c - CIN] - xi[c - CIN];
            return 0.f;
        };
        put_a_pair(Af + (kp >> 5) * 512, m, kp & 31,
                   (_Float16)ev(kp), (_Float16)ev(kp + 1));
    }
    __syncthreads();

    v8f acc = {};
    for (int kc = 0; kc < KC; ++kc) {
        // stage weight chunk in fragment order
        for (int t = tid; t < 16 * NW * 16; t += 32 * NW) {
            int n  = t / 16;           // 0 .. 16*NW-1
            int kp = (t % 16) * 2;     // even klocal
            int gk0 = kc * 32 + kp, gk1 = gk0 + 1;
            float v0 = (gk0 < K2) ? W[(size_t)gk0 * cout + n0 + n] : 0.f;
            float v1 = (gk1 < K2) ? W[(size_t)gk1 * cout + n0 + n] : 0.f;
            put_b_pair(Wf + (n >> 4) * 512, n & 15, kp, (_Float16)v0, (_Float16)v1);
        }
        __syncthreads();
        v16h a   = *(const v16h*)(Af + kc * 512 + lane * 16);
        v16h bfr = *(const v16h*)(Wf + wv * 512 + lane * 16);
        acc = __builtin_amdgcn_wmma_f32_16x16x32_f16(
            false, a, false, bfr, (short)0, acc, false, false);
        __syncthreads();
    }
    {
        const int nc = n0 + wv * 16 + (lane & 15);
        const float bv = bias[nc];
        const int rofs = (lane >> 4) << 3;
        #pragma unroll
        for (int r = 0; r < 8; ++r)
            out[(size_t)(m0 + r + rofs) * cout + nc] = acc[r] + bv;
    }
}

// ---------------------------------------------------------------------------
// Flat row GEMM: out[M, cout] = A[M, Kd] @ W + b (Kd % 32 == 0)
// ---------------------------------------------------------------------------
template<int NW>
__global__ __launch_bounds__(32 * NW) void gemm_rows_kernel(
    const float* __restrict__ A, int lda, int Kd,
    const float* __restrict__ W, const float* __restrict__ bias,
    int cout, float* __restrict__ out)
{
    __shared__ __align__(32) _Float16 Af[512];
    __shared__ __align__(32) _Float16 Wf[NW * 512];
    const int tid = threadIdx.x, wv = tid >> 5, lane = tid & 31;
    const int m0 = blockIdx.x * 16, n0 = blockIdx.y * 16 * NW;

    v8f acc = {};
    for (int k0 = 0; k0 < Kd; k0 += 32) {
        for (int t = tid; t < 16 * 16; t += 32 * NW) {
            int m  = t >> 4;
            int kp = (t & 15) * 2;
            const float* ar = A + (size_t)(m0 + m) * lda + k0 + kp;
            put_a_pair(Af, m, kp, (_Float16)ar[0], (_Float16)ar[1]);
        }
        for (int t = tid; t < 16 * NW * 16; t += 32 * NW) {
            int n  = t / 16;
            int kp = (t % 16) * 2;
            const float* wr = W + (size_t)(k0 + kp) * cout + n0 + n;
            put_b_pair(Wf + (n >> 4) * 512, n & 15, kp,
                       (_Float16)wr[0], (_Float16)wr[cout]);
        }
        __syncthreads();
        v16h a   = *(const v16h*)(Af + lane * 16);
        v16h bfr = *(const v16h*)(Wf + wv * 512 + lane * 16);
        acc = __builtin_amdgcn_wmma_f32_16x16x32_f16(
            false, a, false, bfr, (short)0, acc, false, false);
        __syncthreads();
    }
    const int nc = n0 + wv * 16 + (lane & 15);
    const float bv = bias[nc];
    const int rofs = (lane >> 4) << 3;
    #pragma unroll
    for (int r = 0; r < 8; ++r)
        out[(size_t)(m0 + r + rofs) * cout + nc] = acc[r] + bv;
}

// ---------------------------------------------------------------------------
// BatchNorm stats / finalize / fused BN+SiLU
// ---------------------------------------------------------------------------
__global__ void stats_kernel(const float* __restrict__ x, size_t n, int cols,
                             float* __restrict__ ss)
{
    size_t i0 = (size_t)blockIdx.x * blockDim.x + threadIdx.x;
    size_t stride = (size_t)gridDim.x * blockDim.x;   // stride % cols == 0 by launch
    int c = (int)(i0 % (size_t)cols);
    float s = 0.f, q = 0.f;
    for (size_t i = i0; i < n; i += stride) { float v = x[i]; s += v; q += v * v; }
    atomicAdd(&ss[c], s);
    atomicAdd(&ss[cols + c], q);
}

__global__ void bn_finalize_kernel(const float* __restrict__ ss, int cols, float inv_n,
                                   const float* __restrict__ g, const float* __restrict__ beta,
                                   float* __restrict__ scale, float* __restrict__ shift)
{
    int c = blockIdx.x * blockDim.x + threadIdx.x;
    if (c >= cols) return;
    float m = ss[c] * inv_n;
    float v = ss[cols + c] * inv_n - m * m;
    float s = g[c] * rsqrtf(v + BN_EPS);
    scale[c] = s;
    shift[c] = beta[c] - m * s;
}

__global__ void bn_silu_kernel(const float* __restrict__ src, float* __restrict__ dst,
                               size_t n, int cols, const float* __restrict__ scale,
                               const float* __restrict__ shift)
{
    size_t i = (size_t)blockIdx.x * blockDim.x + threadIdx.x;
    size_t stride = (size_t)gridDim.x * blockDim.x;
    for (; i < n; i += stride) {
        int c = (int)(i % (size_t)cols);
        float y = scale[c] * src[i] + shift[c];
        dst[i] = y / (1.f + __expf(-y));
    }
}

// ---------------------------------------------------------------------------
// Edge gate: gp[e] = h[e,:].Wg + bg, plus global sum/sumsq (1 channel)
// ---------------------------------------------------------------------------
__global__ __launch_bounds__(256) void gate_pre_kernel(
    const float* __restrict__ h, const float* __restrict__ Wg,
    const float* __restrict__ bg, int cout, size_t nedges,
    float* __restrict__ gp, float* __restrict__ ss)
{
    __shared__ float red[256];
    size_t e = (size_t)blockIdx.x * blockDim.x + threadIdx.x;
    float v = 0.f;
    if (e < nedges) {
        const float* hr = h + e * cout;
        float acc = bg[0];
        for (int c = 0; c < cout; ++c) acc += hr[c] * Wg[c];
        gp[e] = acc;
        v = acc;
    }
    red[threadIdx.x] = v; __syncthreads();
    for (int s = 128; s > 0; s >>= 1) {
        if (threadIdx.x < s) red[threadIdx.x] += red[threadIdx.x + s];
        __syncthreads();
    }
    if (threadIdx.x == 0) atomicAdd(&ss[0], red[0]);
    __syncthreads();
    red[threadIdx.x] = v * v; __syncthreads();
    for (int s = 128; s > 0; s >>= 1) {
        if (threadIdx.x < s) red[threadIdx.x] += red[threadIdx.x + s];
        __syncthreads();
    }
    if (threadIdx.x == 0) atomicAdd(&ss[1], red[0]);
}

// ---------------------------------------------------------------------------
// Edge attention combine: softmax over K of silu(bn(gate)), weighted sum of h
// ---------------------------------------------------------------------------
__global__ void attn_combine_kernel(const float* __restrict__ h, const float* __restrict__ gp,
                                    const float* __restrict__ sc, const float* __restrict__ sh,
                                    int cout, float* __restrict__ xc, int ldxc, int colofs)
{
    size_t total = (size_t)BG * NN * cout;
    size_t i = (size_t)blockIdx.x * blockDim.x + threadIdx.x;
    if (i >= total) return;
    size_t node = i / (size_t)cout;
    int c = (int)(i % (size_t)cout);
    float s0 = sc[0], s1 = sh[0];
    float w[8]; float mx = -3.4e38f;
    #pragma unroll
    for (int k = 0; k < 8; ++k) {
        float y = s0 * gp[node * KNB + k] + s1;
        float si = y / (1.f + __expf(-y));
        w[k] = si; mx = fmaxf(mx, si);
    }
    float se = 0.f;
    #pragma unroll
    for (int k = 0; k < 8; ++k) { w[k] = __expf(w[k] - mx); se += w[k]; }
    float acc = 0.f;
    #pragma unroll
    for (int k = 0; k < 8; ++k)
        acc += w[k] * h[(node * KNB + k) * cout + c];
    xc[node * ldxc + colofs + c] = acc / se;
}

// ---------------------------------------------------------------------------
// Node attention pooling: per (graph, channel) softmax over N
// ---------------------------------------------------------------------------
__global__ __launch_bounds__(256) void node_attn_kernel(const float* __restrict__ hs,
                                                        const float* __restrict__ gt,
                                                        float* __restrict__ z)
{
    __shared__ float red[256];
    int b = blockIdx.x >> 9, c = blockIdx.x & 511;
    float gv[8];
    float mx = -3.4e38f;
    #pragma unroll
    for (int k = 0; k < 8; ++k) {
        int n = threadIdx.x + (k << 8);
        gv[k] = gt[((size_t)b * NN + n) * 512 + c];
        mx = fmaxf(mx, gv[k]);
    }
    red[threadIdx.x] = mx; __syncthreads();
    for (int s = 128; s > 0; s >>= 1) {
        if (threadIdx.x < s) red[threadIdx.x] = fmaxf(red[threadIdx.x], red[threadIdx.x + s]);
        __syncthreads();
    }
    mx = red[0]; __syncthreads();

    float se = 0.f, acc = 0.f;
    #pragma unroll
    for (int k = 0; k < 8; ++k) {
        int n = threadIdx.x + (k << 8);
        float e = __expf(gv[k] - mx);
        se += e;
        acc += e * hs[((size_t)b * NN + n) * 512 + c];
    }
    red[threadIdx.x] = se; __syncthreads();
    for (int s = 128; s > 0; s >>= 1) {
        if (threadIdx.x < s) red[threadIdx.x] += red[threadIdx.x + s];
        __syncthreads();
    }
    se = red[0]; __syncthreads();
    red[threadIdx.x] = acc; __syncthreads();
    for (int s = 128; s > 0; s >>= 1) {
        if (threadIdx.x < s) red[threadIdx.x] += red[threadIdx.x + s];
        __syncthreads();
    }
    if (threadIdx.x == 0) z[(size_t)b * 512 + c] = red[0] / se;
}

// ---------------------------------------------------------------------------
// Host orchestration
// ---------------------------------------------------------------------------
extern "C" void kernel_launch(void* const* d_in, const int* in_sizes, int n_in,
                              void* d_out, int out_size, void* d_ws, size_t ws_size,
                              hipStream_t stream)
{
    (void)in_sizes; (void)n_in; (void)out_size; (void)ws_size;
    const float* pos = (const float*)d_in[0];
    auto P = [&](int i) { return (const float*)d_in[i]; };

    const size_t NODES = (size_t)BG * NN;          // 65536
    const size_t EDGES = NODES * KNB;              // 524288

    // workspace layout (bytes)
    char* ws = (char*)d_ws;
    float* xc   = (float*)(ws + 0);                         // 65536 x 256
    float* hE   = (float*)(ws + 67108864);                  // 524288 x 128 (max)
    float* hs   = hE;                                       // reuse: 65536 x 512
    float* gtN  = (float*)(ws + 67108864 + 134217728);      // reuse: 65536 x 512
    float* gp   = (float*)(ws + 335544320);                 // 524288
    int*   idx  = (int*)  (ws + 337641472);                 // 524288
    float* x2   = (float*)(ws + 339738624);                 // 65536
    float* miscF= (float*)(ws + 340000768);
    float* zbuf = (float*)(ws + 340017152);                 // 32 x 512
    float* opre = (float*)(ws + 340082688);                 // 32 x 256

    float* ssA = miscF;            // 2*512
    float* scA = miscF + 1024;     // 512
    float* shA = miscF + 1536;     // 512
    float* ssG = miscF + 2048;     // 2
    float* scG = miscF + 2056;     // 1
    float* shG = miscF + 2060;     // 1

    auto bn_post = [&](float* buf, size_t rows, int cols,
                       const float* g, const float* beta, float* dst) {
        hipMemsetAsync(ssA, 0, 2 * 512 * sizeof(float), stream);
        size_t n = rows * (size_t)cols;
        int sgrid = (rows > 64) ? 512 : 8;   // keeps stride % cols == 0 (cols pow2 <= 512)
        stats_kernel<<<sgrid, 256, 0, stream>>>(buf, n, cols, ssA);
        bn_finalize_kernel<<<1, 512, 0, stream>>>(ssA, cols, 1.f / (float)rows,
                                                  g, beta, scA, shA);
        bn_silu_kernel<<<4096, 256, 0, stream>>>(buf, dst, n, cols, scA, shA);
    };

    auto gate_stage = [&](int cout, const float* Wg, const float* bg,
                          const float* betag, const float* gg, int colofs) {
        hipMemsetAsync(ssG, 0, 2 * sizeof(float), stream);
        gate_pre_kernel<<<(unsigned)(EDGES / 256), 256, 0, stream>>>(
            hE, Wg, bg, cout, EDGES, gp, ssG);
        bn_finalize_kernel<<<1, 32, 0, stream>>>(ssG, 1, 1.f / (float)EDGES,
                                                 gg, betag, scG, shG);
        size_t total = NODES * (size_t)cout;
        attn_combine_kernel<<<(unsigned)((total + 255) / 256), 256, 0, stream>>>(
            hE, gp, scG, shG, cout, xc, 256, colofs);
    };

    // ------------------- conv1: cin=3 (pos), cout=32, out cols [0,32) ------
    rowsq_kernel<<<256, 256, 0, stream>>>(pos, 3, 3, x2, (int)NODES);
    knn_kernel<1, 3><<<dim3(NN / 16, BG), 128, 0, stream>>>(pos, 3, x2, idx);
    edge_gemm_kernel<3, 2><<<dim3((unsigned)(EDGES / 16), 1), 64, 0, stream>>>(
        pos, 3, idx, P(10), P(11), 32, hE);
    bn_post(hE, EDGES, 32, P(13), P(12), hE);
    gate_stage(32, P(6), P(7), P(8), P(9), 0);

    // ------------------- conv2: cin=32 (xc[:,0:32]), cout=32, cols [32,64) -
    rowsq_kernel<<<256, 256, 0, stream>>>(xc + 0, 256, 32, x2, (int)NODES);
    knn_kernel<1, 32><<<dim3(NN / 16, BG), 128, 0, stream>>>(xc + 0, 256, x2, idx);
    edge_gemm_kernel<32, 2><<<dim3((unsigned)(EDGES / 16), 1), 64, 0, stream>>>(
        xc + 0, 256, idx, P(18), P(19), 32, hE);
    bn_post(hE, EDGES, 32, P(21), P(20), hE);
    gate_stage(32, P(14), P(15), P(16), P(17), 32);

    // ------------------- conv3: cin=32 (xc[:,32:64]), cout=64, cols [64,128)
    rowsq_kernel<<<256, 256, 0, stream>>>(xc + 32, 256, 32, x2, (int)NODES);
    knn_kernel<1, 32><<<dim3(NN / 16, BG), 128, 0, stream>>>(xc + 32, 256, x2, idx);
    edge_gemm_kernel<32, 4><<<dim3((unsigned)(EDGES / 16), 1), 128, 0, stream>>>(
        xc + 32, 256, idx, P(26), P(27), 64, hE);
    bn_post(hE, EDGES, 64, P(29), P(28), hE);
    gate_stage(64, P(22), P(23), P(24), P(25), 64);

    // ------------------- conv4: cin=64 (xc[:,64:128]), cout=128, cols [128,256)
    rowsq_kernel<<<256, 256, 0, stream>>>(xc + 64, 256, 64, x2, (int)NODES);
    knn_kernel<2, 64><<<dim3(NN / 16, BG), 128, 0, stream>>>(xc + 64, 256, x2, idx);
    edge_gemm_kernel<64, 4><<<dim3((unsigned)(EDGES / 16), 2), 128, 0, stream>>>(
        xc + 64, 256, idx, P(34), P(35), 128, hE);
    bn_post(hE, EDGES, 128, P(37), P(36), hE);
    gate_stage(128, P(30), P(31), P(32), P(33), 128);

    // ------------------- shared MLP 256 -> 512 ----------------------------
    gemm_rows_kernel<4><<<dim3((unsigned)(NODES / 16), 8), 128, 0, stream>>>(
        xc, 256, 256, P(42), P(43), 512, hs);
    bn_post(hs, NODES, 512, P(45), P(44), hs);

    // ------------------- aggr_gate MLP 512 -> 512 -------------------------
    gemm_rows_kernel<4><<<dim3((unsigned)(NODES / 16), 8), 128, 0, stream>>>(
        hs, 512, 512, P(2), P(3), 512, gtN);
    bn_post(gtN, NODES, 512, P(5), P(4), gtN);

    // ------------------- node attention pooling ---------------------------
    node_attn_kernel<<<BG * 512, 256, 0, stream>>>(hs, gtN, zbuf);

    // ------------------- out MLP 512 -> 256 (BN over B rows) --------------
    gemm_rows_kernel<4><<<dim3(2, 4), 128, 0, stream>>>(
        zbuf, 512, 512, P(38), P(39), 256, opre);
    bn_post(opre, BG, 256, P(41), P(40), (float*)d_out);
}
